// IGCN_67095979098483
// MI455X (gfx1250) — matrix operand order repaired and checked by reading, not compiled, not run on hardware
//
#include <hip/hip_runtime.h>
#include <hip/hip_bf16.h>

// ---------------------------------------------------------------------------
// IGCN forward for MI455X (gfx1250, wave32, WMMA).
//   x_proj = node_features @ W1 + b1          (WMMA bf16 GEMM -> d_ws)
//   out    = node_features[ids] @ W2          (WMMA bf16 GEMM, gathered A)
//   out   += segment_sum(data * x_proj[col], row)   (wave-per-row, no atomics)
// ---------------------------------------------------------------------------

typedef __attribute__((ext_vector_type(16))) __bf16 v16bf;
typedef __attribute__((ext_vector_type(8)))  float  v8f;

#define D_FEAT    512
#define N_CLASSES 64
#define KSTEP     32

__device__ __forceinline__ unsigned short f32_to_bf16_bits(float f) {
    __bf16 h = (__bf16)f;                       // RNE convert (v_cvt)
    return __builtin_bit_cast(unsigned short, h);
}

__device__ __forceinline__ void store_tile(float* __restrict__ Out, v8f acc,
                                           int mbase, int half, int l16,
                                           int nt, const float* __restrict__ bias) {
    const int n  = (nt << 4) + l16;
    const float bv = bias ? bias[n] : 0.0f;
#pragma unroll
    for (int v = 0; v < 8; ++v) {
        // C/D layout: lanes 0-15 -> rows M=0..7 in VGPR0..7, lanes 16-31 -> M=8..15
        Out[(size_t)(mbase + (half << 3) + v) * N_CLASSES + n] = acc[v] + bv;
    }
}

// One wave computes a 16x64 output tile: 4 WMMA accumulators, K-loop of 512/32.
__global__ __launch_bounds__(256)
void igcn_gemm_wmma(const float* __restrict__ A,    // [nNodes, 512] row-major
                    const int*   __restrict__ ids,  // optional row gather
                    const float* __restrict__ W,    // [512, 64] row-major
                    const float* __restrict__ bias, // optional [64]
                    float*       __restrict__ Out,  // [nNodes, 64]
                    int nNodes)
{
    // W transposed to [64][512] bf16 in LDS (64 KB of the 320 KB WGP pool):
    // a B-fragment column becomes 16 contiguous bf16 = 2x ds_load_b128.
    __shared__ __align__(32) unsigned short ldsW[N_CLASSES * D_FEAT];

    for (int i = threadIdx.x; i < D_FEAT * N_CLASSES; i += blockDim.x) {
        const int k = i >> 6;          // 0..511
        const int n = i & 63;          // 0..63
        ldsW[n * D_FEAT + k] = f32_to_bf16_bits(W[i]);
    }
    __syncthreads();

    const int lane          = threadIdx.x & 31;
    const int waveId        = threadIdx.x >> 5;
    const int wavesPerBlock = blockDim.x >> 5;
    const int nWaves        = gridDim.x * wavesPerBlock;
    const int mtiles        = nNodes >> 4;       // 100000/16 = 6250 exact

    const int half = lane >> 4;                  // 0 | 1
    const int l16  = lane & 15;

    for (int tile = blockIdx.x * wavesPerBlock + waveId; tile < mtiles; tile += nWaves) {
        const int mbase = tile << 4;
        int arow = mbase + l16;                  // A 16x32 layout: row = lane&15
        if (ids) arow = ids[arow];
        const float* aptr0 = A + (size_t)arow * D_FEAT + (half << 4);

        v8f acc0 = {}, acc1 = {}, acc2 = {}, acc3 = {};

        for (int kb = 0; kb < D_FEAT; kb += KSTEP) {
            // ---- A fragment (16x32 bf16): lanes 0-15 K 0-15,
            //      lanes 16-31 K 16-31, packed pairs per VGPR.
            const float* ap = aptr0 + kb;
            const float4 f0 = *(const float4*)(ap);
            const float4 f1 = *(const float4*)(ap + 4);
            const float4 f2 = *(const float4*)(ap + 8);
            const float4 f3 = *(const float4*)(ap + 12);
            v16bf a;
            a[0]=(__bf16)f0.x;  a[1]=(__bf16)f0.y;  a[2]=(__bf16)f0.z;  a[3]=(__bf16)f0.w;
            a[4]=(__bf16)f1.x;  a[5]=(__bf16)f1.y;  a[6]=(__bf16)f1.z;  a[7]=(__bf16)f1.w;
            a[8]=(__bf16)f2.x;  a[9]=(__bf16)f2.y;  a[10]=(__bf16)f2.z; a[11]=(__bf16)f2.w;
            a[12]=(__bf16)f3.x; a[13]=(__bf16)f3.y; a[14]=(__bf16)f3.z; a[15]=(__bf16)f3.w;

            // ---- B fragments (32x16 bf16): lane = column, lanes 0-15 K 0-15,
            //      lanes 16-31 K 16-31; 16 contiguous bf16 from transposed LDS.
            const int bko = kb + (half << 4);
            const v16bf b0 = *(const v16bf*)&ldsW[( 0 + l16) * D_FEAT + bko];
            const v16bf b1 = *(const v16bf*)&ldsW[(16 + l16) * D_FEAT + bko];
            const v16bf b2 = *(const v16bf*)&ldsW[(32 + l16) * D_FEAT + bko];
            const v16bf b3 = *(const v16bf*)&ldsW[(48 + l16) * D_FEAT + bko];

            acc0 = __builtin_amdgcn_wmma_f32_16x16x32_bf16(false, a, false, b0, (short)0, acc0, false, false);
            acc1 = __builtin_amdgcn_wmma_f32_16x16x32_bf16(false, a, false, b1, (short)0, acc1, false, false);
            acc2 = __builtin_amdgcn_wmma_f32_16x16x32_bf16(false, a, false, b2, (short)0, acc2, false, false);
            acc3 = __builtin_amdgcn_wmma_f32_16x16x32_bf16(false, a, false, b3, (short)0, acc3, false, false);
        }

        store_tile(Out, acc0, mbase, half, l16, 0, bias);
        store_tile(Out, acc1, mbase, half, l16, 1, bias);
        store_tile(Out, acc2, mbase, half, l16, 2, bias);
        store_tile(Out, acc3, mbase, half, l16, 3, bias);
    }
}

// One wave per output row. Rows are sorted -> binary-search the edge range,
// accumulate in registers (lane owns 2 of 64 columns), one RMW to out.
// x_proj is 25.6 MB -> resident in the 192 MB L2, so gathers hit L2.
__global__ __launch_bounds__(256)
void igcn_spmm_rows(const int*   __restrict__ row,
                    const int*   __restrict__ col,
                    const float* __restrict__ dat,
                    const float* __restrict__ xproj, // [nNodes, 64]
                    float*       __restrict__ out,   // [nNodes, 64], pre-filled
                    int nNodes, int nEdges)
{
    const int wave = (int)((blockIdx.x * blockDim.x + threadIdx.x) >> 5);
    const int lane = threadIdx.x & 31;
    if (wave >= nNodes) return;
    const int r = wave;

    // lower_bound(row, r) and lower_bound(row, r+1) — wave-uniform scalar work
    int lo = 0, hi = nEdges;
    while (lo < hi) { const int mid = (lo + hi) >> 1; if (row[mid] < r) lo = mid + 1; else hi = mid; }
    const int start = lo;
    hi = nEdges;
    while (lo < hi) { const int mid = (lo + hi) >> 1; if (row[mid] < r + 1) lo = mid + 1; else hi = mid; }
    const int end = lo;

    float ax = 0.0f, ay = 0.0f;
    for (int e = start; e < end; ++e) {
        const float v = dat[e];
        const int   c = col[e];
        const float2 xv = *(const float2*)(xproj + (size_t)c * N_CLASSES + (lane << 1));
        ax = fmaf(v, xv.x, ax);
        ay = fmaf(v, xv.y, ay);
    }

    float2* o = (float2*)(out + (size_t)r * N_CLASSES + (lane << 1));
    float2 cur = *o;
    cur.x += ax; cur.y += ay;
    *o = cur;
}

extern "C" void kernel_launch(void* const* d_in, const int* in_sizes, int n_in,
                              void* d_out, int out_size, void* d_ws, size_t ws_size,
                              hipStream_t stream) {
    const float* nf  = (const float*)d_in[0];   // [nNodes, 512]
    const int*   row = (const int*)  d_in[1];   // [nEdges]
    const int*   col = (const int*)  d_in[2];   // [nEdges]
    const float* dat = (const float*)d_in[3];   // [nEdges]
    const int*   ids = (const int*)  d_in[4];   // [nNodes]
    const float* W1  = (const float*)d_in[5];   // [512, 64]
    const float* b1  = (const float*)d_in[6];   // [64]
    const float* W2  = (const float*)d_in[7];   // [512, 64]

    float* out   = (float*)d_out;
    float* xproj = (float*)d_ws;                // [nNodes, 64] scratch (25.6 MB)

    const int nNodes = in_sizes[0] / D_FEAT;    // 100000
    const int nEdges = in_sizes[1];             // 1600000

    // GEMM1: x_proj = nf @ W1 + b1
    igcn_gemm_wmma<<<640, 256, 0, stream>>>(nf, nullptr, W1, b1, xproj, nNodes);
    // GEMM2: out = nf[ids] @ W2   (initializes out)
    igcn_gemm_wmma<<<640, 256, 0, stream>>>(nf, ids, W2, nullptr, out, nNodes);
    // SpMM: out += segment_sum(data * x_proj[col], row)
    const int spmmBlocks = (nNodes * 32 + 255) / 256;   // one wave per row
    igcn_spmm_rows<<<spmmBlocks, 256, 0, stream>>>(row, col, dat, xproj, out, nNodes, nEdges);
}
